// SelfAttention_47940424958431
// MI455X (gfx1250) — matrix operand order repaired and checked by reading, not compiled, not run on hardware
//
#include <hip/hip_runtime.h>
#include <hip/hip_bf16.h>

typedef __attribute__((ext_vector_type(16))) _Float16 v16h;
typedef __attribute__((ext_vector_type(4)))  _Float16 v4h;
typedef __attribute__((ext_vector_type(8)))  float    v8f;
typedef __attribute__((ext_vector_type(4)))  unsigned int u32x4;
typedef __attribute__((ext_vector_type(8)))  int          i32x8;
typedef __attribute__((ext_vector_type(4)))  int          i32x4;

#define HID 1024
#define SEQ 2048
#define BATCH 8

#if defined(__has_builtin)
#if __has_builtin(__builtin_amdgcn_tensor_load_to_lds) && __has_builtin(__builtin_amdgcn_s_wait_tensorcnt)
#define USE_TDM 1
#endif
#endif

// ---------------------------------------------------------------------------
// WMMA helper (CDNA5 wave32, V_WMMA_F32_16X16X32_F16)
// ---------------------------------------------------------------------------
__device__ inline v8f wmma_f16(v16h a, v16h b, v8f c) {
  return __builtin_amdgcn_wmma_f32_16x16x32_f16(
      /*neg_a=*/false, a, /*neg_b=*/false, b,
      /*c_mod=*/(short)0, c, /*reuse_a=*/false, /*reuse_b=*/false);
}

// A-matrix 16x32 f16 from row-major storage (row stride ld elements).
// Lane L: row r=L%16; koff=16*(L/16); elems {koff..koff+15}.
__device__ inline v16h load_a_rowmajor(const _Float16* p, int ld, int lane) {
  int r    = lane & 15;
  int koff = (lane >> 4) << 4;
  const _Float16* row = p + (size_t)r * ld;
  v16h a;
#pragma unroll
  for (int i = 0; i < 8; ++i) a[i] = row[koff + i];
#pragma unroll
  for (int i = 0; i < 8; ++i) a[8 + i] = row[koff + 8 + i];
  return a;
}

// B-matrix 32x16: B[k][n] = src[n*ldt + k] (column n contiguous in k,
// one 32-byte load per lane).
__device__ inline v16h load_b_coltile(const _Float16* src, int ldt, int lane) {
  int n     = lane & 15;
  int kbase = (lane >> 4) << 4;
  const _Float16* col = src + (size_t)n * ldt + kbase;
  v16h b;
#pragma unroll
  for (int i = 0; i < 16; ++i) b[i] = col[i];
  return b;
}

// ---------------------------------------------------------------------------
// TDM: 1-D contiguous tile (nelem f16 elements) global -> LDS.
// D# per CDNA5 ISA 8.3/8.4: group0 {count=1, lds_addr, global_addr, type=2};
// group1 {data_size=2B, tensor_dim0=nelem, tensor_dim1=1, tile_dim0=nelem,
// dim0_stride=nelem}.  Issued by wave 0; drained with s_wait_tensorcnt 0.
// ---------------------------------------------------------------------------
__device__ inline void stage_tile_lds(_Float16* lds, const _Float16* gsrc,
                                      unsigned nelem, int wid, int tid,
                                      int nthreads) {
#ifdef USE_TDM
  if (wid == 0) {
    unsigned long long ga = (unsigned long long)(uintptr_t)gsrc;
    unsigned lo = (unsigned)(uintptr_t)lds;
    u32x4 g0;
    g0[0] = 1u;                                            // count=1, user D#
    g0[1] = lo;                                            // lds_addr
    g0[2] = (unsigned)(ga & 0xFFFFFFFFu);                  // global_addr[31:0]
    g0[3] = (unsigned)((ga >> 32) & 0x01FFFFFFu) | 0x80000000u;  // addr[56:32] | type=2
    i32x8 g1;
    g1[0] = 0x00010000;                 // data_size=1 (2 bytes)
    g1[1] = (int)(nelem << 16);         // tensor_dim0[15:0]
    g1[2] = (int)((nelem >> 16) & 0xFFFFu) | (1 << 16);  // dim0[31:16] | tensor_dim1=1
    g1[3] = (int)((nelem & 0xFFFFu) << 16);              // tile_dim0 (1-D tile)
    g1[4] = 0;                          // tile_dim1=0 (unused), tile_dim2=0
    g1[5] = (int)nelem;                 // tensor_dim0_stride[31:0]
    g1[6] = 0;
    g1[7] = 0;
    i32x4 gz = {0, 0, 0, 0};
#if __clang_major__ >= 23
    i32x8 gz8 = {0, 0, 0, 0, 0, 0, 0, 0};
    __builtin_amdgcn_tensor_load_to_lds(g0, g1, gz, gz, gz8, 0);
#else
    __builtin_amdgcn_tensor_load_to_lds(g0, g1, gz, gz, 0);
#endif
    __builtin_amdgcn_s_wait_tensorcnt(0);
  }
#else
  const uint4* s = (const uint4*)gsrc;
  uint4* d = (uint4*)lds;
  for (unsigned i = tid; i < nelem / 8; i += nthreads) d[i] = s[i];
#endif
  (void)tid; (void)nthreads; (void)wid;
}

// ---------------------------------------------------------------------------
// Prep kernels
// ---------------------------------------------------------------------------
__global__ __launch_bounds__(256) void cvt_f32_to_f16_k(const float* __restrict__ src,
                                                        _Float16* __restrict__ dst,
                                                        int n4) {
  int i = blockIdx.x * 256 + threadIdx.x;
  if (i < n4) {
    float4 f = ((const float4*)src)[i];
    v4h h = {(_Float16)f.x, (_Float16)f.y, (_Float16)f.z, (_Float16)f.w};
    ((v4h*)dst)[i] = h;
  }
}

// WT[n][k] = (f16) W[k][n]
__global__ __launch_bounds__(256) void transpose_w_k(const float* __restrict__ W,
                                                     _Float16* __restrict__ WT,
                                                     int K, int N) {
  __shared__ float tile[16][17];
  int tx = threadIdx.x & 15, ty = threadIdx.x >> 4;
  int k0 = blockIdx.y * 16, n0 = blockIdx.x * 16;
  tile[ty][tx] = W[(size_t)(k0 + ty) * N + n0 + tx];
  __syncthreads();
  WT[(size_t)(n0 + ty) * K + k0 + tx] = (_Float16)tile[tx][ty];
}

// ---------------------------------------------------------------------------
// GEMM: C = scale * (A[M][1024] @ W + bias)
// One block = one 16-row M strip; A strip (32 KB) staged once in LDS via TDM
// and shared by 8 waves; wave w owns N slice [w*128, w*128+128) -> 8 acc
// tiles, A fragment reused across 8 WMMAs per 32-wide k-step.
// MODE: 0 = f16 row-major [M][1024]
//       1 = f16 transposed per batch: VT[b][n][t] (n in [0,1024), t in [0,SEQ))
//       2 = f32 row-major [M][1024]
// grid = (M/16), block = 256
// ---------------------------------------------------------------------------
template <int MODE>
__global__ __launch_bounds__(256) void gemm_strip16_k(const _Float16* __restrict__ A,
                                                      const _Float16* __restrict__ BT,
                                                      const float* __restrict__ bias,
                                                      void* __restrict__ Cv,
                                                      float scale) {
  __shared__ _Float16 As[16 * HID];  // 32 KB
  int lane = threadIdx.x & 31, wid = threadIdx.x >> 5;
  size_t mbase = (size_t)blockIdx.x * 16;

  stage_tile_lds(As, A + mbase * HID, 16 * HID, wid, threadIdx.x, 256);
  __syncthreads();

  int nbase = wid * 128;
  v8f acc[8];
#pragma unroll
  for (int t = 0; t < 8; ++t) acc[t] = v8f{0.f, 0.f, 0.f, 0.f, 0.f, 0.f, 0.f, 0.f};

  for (int k0 = 0; k0 < HID; k0 += 32) {
    v16h a = load_a_rowmajor(As + k0, HID, lane);
#pragma unroll
    for (int t = 0; t < 8; ++t) {
      v16h b = load_b_coltile(BT + (size_t)(nbase + t * 16) * HID + k0, HID, lane);
      acc[t] = wmma_f16(a, b, acc[t]);
    }
  }

  int rb = (lane >> 4) * 8;
#pragma unroll
  for (int t = 0; t < 8; ++t) {
    int n = nbase + t * 16 + (lane & 15);
    float bv = bias[n];
    if (MODE == 2) {
      float* Cp = (float*)Cv + mbase * HID + n;
#pragma unroll
      for (int v = 0; v < 8; ++v)
        Cp[(size_t)(rb + v) * HID] = acc[t][v] + bv;
    } else if (MODE == 1) {
      // transposed: rows m = mbase+rb+v map to consecutive t positions ->
      // one contiguous 16-byte store per lane per tile
      int bb = (int)(mbase / SEQ);
      int tbase = (int)(mbase % SEQ);
      _Float16* dst = (_Float16*)Cv + ((size_t)bb * HID + n) * SEQ + tbase + rb;
#pragma unroll
      for (int v = 0; v < 8; ++v)
        dst[v] = (_Float16)((acc[t][v] + bv) * scale);
    } else {
      _Float16* Cp = (_Float16*)Cv + mbase * HID + n;
#pragma unroll
      for (int v = 0; v < 8; ++v)
        Cp[(size_t)(rb + v) * HID] = (_Float16)((acc[t][v] + bv) * scale);
    }
  }
}

// ---------------------------------------------------------------------------
// Flash attention (causal), Q pre-scaled by 1/sqrt(dk).
// One workgroup (8 waves) per (batch, 16-row query tile); Q tile staged via
// TDM.  dk split across waves for S (ds_add_f32 reduction in LDS), wave 0
// runs the online softmax, dv split across waves for ctx += P.V with the
// B operand coming from the pre-transposed VT (contiguous 32B per lane).
// grid = (SEQ/16, BATCH), block = 256
// ---------------------------------------------------------------------------
__global__ __launch_bounds__(256) void attn_flash_k(const _Float16* __restrict__ Q,
                                                    const _Float16* __restrict__ Km,
                                                    const _Float16* __restrict__ VT,
                                                    _Float16* __restrict__ CTX) {
  __shared__ _Float16 Qs[16 * HID];   // 32 KB
  __shared__ float    Ss[16 * 32];    // 2 KB
  __shared__ _Float16 Ps[16 * 32];    // 1 KB
  __shared__ float    Ms[16], Ls[16], Al[16];

  int lane = threadIdx.x & 31, wid = threadIdx.x >> 5;
  int b = blockIdx.y;
  int qbase = blockIdx.x * 16;

  const _Float16* Qb  = Q + ((size_t)b * SEQ + qbase) * HID;
  const _Float16* Kb  = Km + (size_t)b * SEQ * HID;
  const _Float16* VTb = VT + (size_t)b * HID * SEQ;  // [dv][t]

  stage_tile_lds(Qs, Qb, 16 * HID, wid, threadIdx.x, 256);
  if (threadIdx.x < 16) { Ms[threadIdx.x] = -1e30f; Ls[threadIdx.x] = 0.f; }
  __syncthreads();

  v8f acc[8];
#pragma unroll
  for (int t = 0; t < 8; ++t) acc[t] = v8f{0.f, 0.f, 0.f, 0.f, 0.f, 0.f, 0.f, 0.f};

  int dkbase = wid * 128;
  int dvbase = wid * 128;
  int nblk = (qbase + 47) >> 5;  // 32-wide key blocks covering keys [0, qbase+16)

  for (int jb = 0; jb < nblk; ++jb) {
    int jbase = jb * 32;

    // hide latency of the next key block behind this block's WMMA work
    if (jb + 1 < nblk) {
      const char* nk = (const char*)(Kb + (size_t)(jbase + 32) * HID);
      __builtin_prefetch(nk + threadIdx.x * 256, 0, 1);
      const char* nv = (const char*)(VTb + (size_t)(threadIdx.x * 4) * SEQ + jbase + 32);
      __builtin_prefetch(nv, 0, 1);
    }

    for (int i = threadIdx.x; i < 512; i += 256) Ss[i] = 0.f;
    __syncthreads();

    // --- partial S = Q_slice . K_slice^T (two 16x16 n-halves) ---
    v8f s0 = {0.f, 0.f, 0.f, 0.f, 0.f, 0.f, 0.f, 0.f};
    v8f s1 = {0.f, 0.f, 0.f, 0.f, 0.f, 0.f, 0.f, 0.f};
#pragma unroll
    for (int kk = 0; kk < 128; kk += 32) {
      v16h a  = load_a_rowmajor(Qs + dkbase + kk, HID, lane);
      v16h b0 = load_b_coltile(Kb + (size_t)jbase * HID + dkbase + kk, HID, lane);
      v16h b1 = load_b_coltile(Kb + (size_t)(jbase + 16) * HID + dkbase + kk, HID, lane);
      s0 = wmma_f16(a, b0, s0);
      s1 = wmma_f16(a, b1, s1);
    }
    {
      int n = lane & 15, rb = (lane >> 4) * 8;
#pragma unroll
      for (int v = 0; v < 8; ++v) {
        atomicAdd(&Ss[(rb + v) * 32 + n], s0[v]);
        atomicAdd(&Ss[(rb + v) * 32 + 16 + n], s1[v]);
      }
    }
    __syncthreads();

    // --- online softmax on the 16x32 S tile (wave 0, one lane per row) ---
    if (wid == 0 && lane < 16) {
      int r = lane, qi = qbase + r;
      float mv = Ms[r], lv = Ls[r];
      float x[32];
      float rowmax = mv;
#pragma unroll
      for (int c = 0; c < 32; ++c) {
        float s = Ss[r * 32 + c];
        s = (jbase + c <= qi) ? s : -1e30f;
        x[c] = s;
        rowmax = fmaxf(rowmax, s);
      }
      float alpha = __expf(mv - rowmax);
      float sum = lv * alpha;
#pragma unroll
      for (int c = 0; c < 32; ++c) {
        float p = __expf(x[c] - rowmax);
        Ps[r * 32 + c] = (_Float16)p;
        sum += p;
      }
      Ms[r] = rowmax;
      Ls[r] = sum;
      Al[r] = alpha;
    }
    __syncthreads();

    // --- rescale ctx and accumulate P.V for this wave's dv slice ---
    {
      int rb = (lane >> 4) * 8;
      float al[8];
#pragma unroll
      for (int v = 0; v < 8; ++v) al[v] = Al[rb + v];
      v16h a = load_a_rowmajor(Ps, 32, lane);
#pragma unroll
      for (int t = 0; t < 8; ++t) {
#pragma unroll
        for (int v = 0; v < 8; ++v) acc[t][v] *= al[v];
        v16h bm = load_b_coltile(VTb + (size_t)(dvbase + t * 16) * SEQ + jbase, SEQ, lane);
        acc[t] = wmma_f16(a, bm, acc[t]);
      }
    }
    __syncthreads();
  }

  // --- finalize: ctx /= l, write f16 row-major for the output GEMM ---
  {
    int n = lane & 15, rb = (lane >> 4) * 8;
    float linv[8];
#pragma unroll
    for (int v = 0; v < 8; ++v) linv[v] = 1.0f / Ls[rb + v];
    _Float16* Cb = CTX + ((size_t)b * SEQ + qbase) * HID + dvbase;
#pragma unroll
    for (int t = 0; t < 8; ++t)
#pragma unroll
      for (int v = 0; v < 8; ++v)
        Cb[(size_t)(rb + v) * HID + t * 16 + n] = (_Float16)(acc[t][v] * linv[v]);
  }
}

// ---------------------------------------------------------------------------
// Launch
// ---------------------------------------------------------------------------
extern "C" void kernel_launch(void* const* d_in, const int* in_sizes, int n_in,
                              void* d_out, int out_size, void* d_ws, size_t ws_size,
                              hipStream_t stream) {
  (void)in_sizes; (void)n_in; (void)out_size; (void)ws_size;

  const float* y  = (const float*)d_in[0];
  const float* Wq = (const float*)d_in[1];
  const float* bq = (const float*)d_in[2];
  const float* Wk = (const float*)d_in[3];
  const float* bk = (const float*)d_in[4];
  const float* Wv = (const float*)d_in[5];
  const float* bv = (const float*)d_in[6];
  const float* Wo = (const float*)d_in[7];
  const float* bo = (const float*)d_in[8];
  float* out = (float*)d_out;

  const size_t M = (size_t)BATCH * SEQ;            // 16384
  const size_t big = M * HID * sizeof(_Float16);   // 32 MB each
  const size_t wsz = (size_t)HID * HID * sizeof(_Float16);

  char* ws = (char*)d_ws;
  _Float16* yh   = (_Float16*)(ws);
  _Float16* Qh   = (_Float16*)(ws + big);
  _Float16* Kh   = (_Float16*)(ws + 2 * big);
  _Float16* VTh  = (_Float16*)(ws + 3 * big);      // per-batch [HID][SEQ]
  _Float16* CTXh = (_Float16*)(ws + 4 * big);
  _Float16* WqT  = (_Float16*)(ws + 5 * big);
  _Float16* WkT  = (_Float16*)(ws + 5 * big + wsz);
  _Float16* WvT  = (_Float16*)(ws + 5 * big + 2 * wsz);
  _Float16* WoT  = (_Float16*)(ws + 5 * big + 3 * wsz);

  // prep: fp32 -> f16 conversion / weight transposes
  cvt_f32_to_f16_k<<<(int)(M * HID / 4 / 256), 256, 0, stream>>>(y, yh, (int)(M * HID / 4));
  dim3 tg(HID / 16, HID / 16);
  transpose_w_k<<<tg, 256, 0, stream>>>(Wq, WqT, HID, HID);
  transpose_w_k<<<tg, 256, 0, stream>>>(Wk, WkT, HID, HID);
  transpose_w_k<<<tg, 256, 0, stream>>>(Wv, WvT, HID, HID);
  transpose_w_k<<<tg, 256, 0, stream>>>(Wo, WoT, HID, HID);

  // QKV projections (Q folded with 1/sqrt(dk) = 1/32; V written transposed)
  dim3 pg((unsigned)(M / 16));
  gemm_strip16_k<0><<<pg, 256, 0, stream>>>(yh, WqT, bq, Qh, 0.03125f);
  gemm_strip16_k<0><<<pg, 256, 0, stream>>>(yh, WkT, bk, Kh, 1.0f);
  gemm_strip16_k<1><<<pg, 256, 0, stream>>>(yh, WvT, bv, VTh, 1.0f);

  // causal flash attention
  dim3 ag(SEQ / 16, BATCH);
  attn_flash_k<<<ag, 256, 0, stream>>>(Qh, Kh, VTh, CTXh);

  // output projection -> fp32
  gemm_strip16_k<2><<<pg, 256, 0, stream>>>(CTXh, WoT, bo, out, 1.0f);
}
